// GCN_drop_30202210026006
// MI455X (gfx1250) — compile-verified, matrix operand order checked by reading, not run on hardware
//
#include <hip/hip_runtime.h>
#include <hip/hip_bf16.h>
#include <math.h>

typedef __attribute__((ext_vector_type(16))) _Float16 v16h;
typedef __attribute__((ext_vector_type(8)))  _Float16 v8h;
typedef __attribute__((ext_vector_type(4)))  _Float16 v4h;
typedef __attribute__((ext_vector_type(8)))  float    v8f;

// ---------------------------------------------------------------- utilities
__global__ void fill0_kernel(float* __restrict__ p, long n) {
    long i = (long)blockIdx.x * blockDim.x + threadIdx.x;
    if (i < n) p[i] = 0.0f;
}

__global__ void deg_kernel(const int* __restrict__ src, const int* __restrict__ dst,
                           float* __restrict__ outdeg, float* __restrict__ indeg, int E) {
    int e = blockIdx.x * blockDim.x + threadIdx.x;
    if (e < E) {
        atomicAdd(&outdeg[src[e]], 1.0f);
        atomicAdd(&indeg[dst[e]], 1.0f);
    }
}

__global__ void invsqrt_kernel(float* __restrict__ p, long n) {
    long i = (long)blockIdx.x * blockDim.x + threadIdx.x;
    if (i < n) {
        float d = p[i];
        p[i] = (d > 0.0f) ? (1.0f / sqrtf(d)) : 0.0f;
    }
}

// Pre-swizzle row-major W[K][NCsrc] (fp32) into per-lane WMMA-B fragments (f16).
// Packed layout: [kt][nt][lane][e]   (e = 16 contiguous f16 per lane = 32 bytes)
// element e of lane L  ->  kk = (e<8 ? e : e+8) + (L<16 ? 0 : 8),  n = L&15
__global__ void packB_kernel(const float* __restrict__ W, _Float16* __restrict__ Bp,
                             int Ksrc, int NCsrc, int NTn, int total) {
    int idx = blockIdx.x * blockDim.x + threadIdx.x;
    if (idx >= total) return;
    int e  = idx & 15;
    int L  = (idx >> 4) & 31;
    int nt = (idx >> 9) % NTn;
    int kt = idx / (512 * NTn);
    int kk = (e < 8 ? e : e + 8) + (L < 16 ? 0 : 8);
    int n  = nt * 16 + (L & 15);
    int k  = kt * 32 + kk;
    float v = (n < NCsrc && k < Ksrc) ? W[(long)k * NCsrc + n] : 0.0f;
    Bp[idx] = (_Float16)v;
}

// ------------------------------------------------------------- WMMA GEMM
// Out[M x (NT*16)] = (A[M x (KT*32)] * rs[row]) @ Bp   (f16 mul, f32 accum)
// 128 threads = 4 wave32; block covers 64 rows; wave w covers rows w*16..+15.
// Double-buffered LDS A tile; depth-2 pipelined B-fragment loads (low VGPR pressure).
template <int KT, int NT>
__global__ void gemm_wmma_kernel(const float* __restrict__ A,
                                 const _Float16* __restrict__ Bp,
                                 const float* __restrict__ rs,
                                 float* __restrict__ Out,
                                 int M, int lda, int ldc) {
    __shared__ __align__(16) _Float16 As[2][64 * 32];

    const int tid  = threadIdx.x;
    const int lane = tid & 31;
    const int w    = tid >> 5;
    const int rowBase = blockIdx.x * 64;
    const int hi   = (lane < 16) ? 0 : 8;   // K-half / M-half selector
    const int mrow = lane & 15;

    // stage one K-tile: 64 rows x 32 k, float4 loads -> f16x4 LDS stores.
    // rows clamped: out-of-range rows feed only store-guarded outputs.
    auto stage = [&](int kt, int buf) {
        #pragma unroll
        for (int j = 0; j < 4; ++j) {
            int base = j * 512 + tid * 4;          // element index in 64x32 tile
            int r  = base >> 5;
            int kk = base & 31;
            int gr = rowBase + r;
            if (gr >= M) gr = M - 1;
            const float4 v = *(const float4*)(A + (long)gr * lda + kt * 32 + kk);
            const float  s = rs[gr];
            v4h h;
            h[0] = (_Float16)(v.x * s);
            h[1] = (_Float16)(v.y * s);
            h[2] = (_Float16)(v.z * s);
            h[3] = (_Float16)(v.w * s);
            *(v4h*)&As[buf][base] = h;
        }
    };

    v8f acc[NT] = {};
    stage(0, 0);

    for (int kt = 0; kt < KT; ++kt) {
        const int buf = kt & 1;
        __syncthreads();   // staging of buf complete; prior reads of buf^1 done

        // A fragment: two 16B LDS loads per lane
        union { v16h v; v8h h[2]; } af;
        const _Float16* ap = &As[buf][(w * 16 + mrow) * 32];
        af.h[0] = *(const v8h*)(ap + hi);
        af.h[1] = *(const v8h*)(ap + 16 + hi);

        // overlap next tile's staging with this K-step's B loads / WMMAs
        if (kt + 1 < KT) stage(kt + 1, buf ^ 1);

        // depth-2 pipelined B fragments: only 2 live at a time
        const _Float16* bbase = Bp + (long)kt * NT * 512 + (long)lane * 16;
        v16h bcur = *(const v16h*)(bbase);
        #pragma unroll
        for (int nt = 0; nt < NT; ++nt) {
            v16h bnext = bcur;
            if (nt + 1 < NT) bnext = *(const v16h*)(bbase + (nt + 1) * 512);
            acc[nt] = __builtin_amdgcn_wmma_f32_16x16x32_f16(
                false, af.v, false, bcur, (short)0, acc[nt], false, false);
            bcur = bnext;
        }
    }

    // D layout: lane -> col (lane&15), VGPR r -> row (r + hi)
    const int col0 = lane & 15;
    #pragma unroll
    for (int nt = 0; nt < NT; ++nt) {
        #pragma unroll
        for (int r = 0; r < 8; ++r) {
            int orow = rowBase + w * 16 + hi + r;
            if (orow < M) Out[(long)orow * ldc + nt * 16 + col0] = acc[nt][r];
        }
    }
}

// ---------------------------------------------------------- edge scatter
// one thread per (edge, 4-col group): float4 gather + 4 f32 atomics
__global__ void scatter4_kernel(const int* __restrict__ src, const int* __restrict__ dst,
                                const float* __restrict__ h, float* __restrict__ agg,
                                int E, int C4, int ldh, int ldo) {
    long i = (long)blockIdx.x * blockDim.x + threadIdx.x;
    long total = (long)E * C4;
    if (i >= total) return;
    int e = (int)(i / C4);
    int c = (int)(i % C4) * 4;
    int s = src[e];
    int d = dst[e];
    const float4 v = *(const float4*)(h + (long)s * ldh + c);
    float* o = agg + (long)d * ldo + c;
    atomicAdd(o + 0, v.x);
    atomicAdd(o + 1, v.y);
    atomicAdd(o + 2, v.z);
    atomicAdd(o + 3, v.w);
}

// x[i] = (x[i]*norm_dst[row] + b[col])  [optionally ReLU'd], in place
__global__ void bias_norm_kernel(float* __restrict__ x, const float* __restrict__ nd,
                                 const float* __restrict__ b, long total, int C, int doRelu) {
    long i = (long)blockIdx.x * blockDim.x + threadIdx.x;
    if (i >= total) return;
    int row = (int)(i / C);
    int c   = (int)(i % C);
    float v = x[i] * nd[row] + b[c];
    x[i] = doRelu ? fmaxf(v, 0.0f) : v;
}

// ---------------------------------------------------------------- launch
extern "C" void kernel_launch(void* const* d_in, const int* in_sizes, int n_in,
                              void* d_out, int out_size, void* d_ws, size_t ws_size,
                              hipStream_t stream) {
    const float* features = (const float*)d_in[0];
    const int*   src      = (const int*)d_in[1];
    const int*   dst      = (const int*)d_in[2];
    const float* W1       = (const float*)d_in[3];
    const float* b1       = (const float*)d_in[4];
    const float* W2       = (const float*)d_in[5];
    const float* b2       = (const float*)d_in[6];
    float*       out      = (float*)d_out;

    const int FIN = 256, NH = 128, NCLS = 40, NCP = 48;
    const int Nn = in_sizes[0] / FIN;     // 100000
    const int E  = in_sizes[1];           // 800000

    // workspace carve (256B aligned)
    size_t off = 0;
    auto carve = [&](size_t bytes) -> char* {
        char* p = (char*)d_ws + off;
        off += (bytes + 255) & ~(size_t)255;
        return p;
    };
    float*    norm_src = (float*)carve((size_t)2 * Nn * sizeof(float)); // [2N]: src then dst
    float*    norm_dst = norm_src + Nn;
    _Float16* W1p      = (_Float16*)carve((size_t)8 * 8 * 512 * sizeof(_Float16));
    _Float16* W2p      = (_Float16*)carve((size_t)4 * 3 * 512 * sizeof(_Float16));
    float*    h1       = (float*)carve((size_t)Nn * NH * sizeof(float)); // reused as h2
    float*    x1       = (float*)carve((size_t)Nn * NH * sizeof(float)); // agg1 -> x1 in place

    const int T = 256;
    auto blocks = [](long n, int t) { return (unsigned)((n + t - 1) / t); };

    // --- degree norms ---
    fill0_kernel<<<blocks((long)2 * Nn, T), T, 0, stream>>>(norm_src, (long)2 * Nn);
    deg_kernel<<<blocks(E, T), T, 0, stream>>>(src, dst, norm_src, norm_dst, E);
    invsqrt_kernel<<<blocks((long)2 * Nn, T), T, 0, stream>>>(norm_src, (long)2 * Nn);

    // --- pack weights into WMMA-B fragment layout ---
    packB_kernel<<<blocks(8 * 8 * 512, T), T, 0, stream>>>(W1, W1p, FIN, NH, 8, 8 * 8 * 512);
    packB_kernel<<<blocks(4 * 3 * 512, T), T, 0, stream>>>(W2, W2p, NH, NCLS, 3, 4 * 3 * 512);

    // --- layer 1 ---
    gemm_wmma_kernel<8, 8><<<blocks(Nn, 64), 128, 0, stream>>>(
        features, W1p, norm_src, h1, Nn, FIN, NH);
    fill0_kernel<<<blocks((long)Nn * NH, T), T, 0, stream>>>(x1, (long)Nn * NH);
    scatter4_kernel<<<blocks((long)E * (NH / 4), T), T, 0, stream>>>(
        src, dst, h1, x1, E, NH / 4, NH, NH);
    bias_norm_kernel<<<blocks((long)Nn * NH, T), T, 0, stream>>>(
        x1, norm_dst, b1, (long)Nn * NH, NH, 1);

    // --- layer 2 --- (h2 reuses h1 buffer, padded to 48 cols)
    gemm_wmma_kernel<4, 3><<<blocks(Nn, 64), 128, 0, stream>>>(
        x1, W2p, norm_src, h1, Nn, NH, NCP);
    fill0_kernel<<<blocks((long)Nn * NCLS, T), T, 0, stream>>>(out, (long)Nn * NCLS);
    scatter4_kernel<<<blocks((long)E * (NCLS / 4), T), T, 0, stream>>>(
        src, dst, h1, out, E, NCLS / 4, NCP, NCLS);
    bias_norm_kernel<<<blocks((long)Nn * NCLS, T), T, 0, stream>>>(
        out, norm_dst, b2, (long)Nn * NCLS, NCLS, 0);
}
